// LinearSqrtCrossEntropyLoss_70300024701270
// MI455X (gfx1250) — matrix operand order repaired and checked by reading, not compiled, not run on hardware
//
#include <hip/hip_runtime.h>

#define IGNORE_INDEX (-100)
#define B_  2
#define S_  4096
#define D_  1024
#define V_  32000
#define NROWS (B_ * S_)          // 8192 tokens
#define VCHUNK 256               // columns per block; 32000 / 256 = 125 exactly
#define NVC (V_ / VCHUNK)        // 125
#define MTILE 64                 // rows per block (4 WMMA M-tiles per wave)
#define KC 128                   // K-chunk staged in LDS
#define LDS_STRIDE (KC + 8)      // 136 bf16: row stride padding vs. bank conflicts

typedef __attribute__((ext_vector_type(16))) __bf16 v16bf;
typedef __attribute__((ext_vector_type(8)))  float  v8f;

// ---------- f32 -> bf16 round-to-nearest-even ----------
__device__ __forceinline__ unsigned short f2bf(float f) {
    unsigned int u = __float_as_uint(f);
    u += 0x7FFFu + ((u >> 16) & 1u);
    return (unsigned short)(u >> 16);
}

// ---------- kernel 1: convert X/W to bf16, zero target-logit buffer ----------
__global__ void prep_kernel(const float* __restrict__ X, const float* __restrict__ W,
                            unsigned short* __restrict__ Xb, unsigned short* __restrict__ Wb,
                            float* __restrict__ tgtLogit, long nX, long nW, int nRows) {
    long i = (long)blockIdx.x * blockDim.x + threadIdx.x;
    long stride = (long)gridDim.x * blockDim.x;
    for (long k = i; k < nW; k += stride) Wb[k] = f2bf(W[k]);
    for (long k = i; k < nX; k += stride) Xb[k] = f2bf(X[k]);
    for (long k = i; k < nRows; k += stride) tgtLogit[k] = 0.0f;
}

// ---------- kernel 2: LDS-staged bf16 WMMA GEMM + per-chunk softmax partials ----------
__global__ __launch_bounds__(256)
void gemm_partial_kernel(const unsigned short* __restrict__ Xb,
                         const unsigned short* __restrict__ Wb,
                         const int* __restrict__ targets,
                         float* __restrict__ pMax,
                         float* __restrict__ pSum,
                         float* __restrict__ tgtLogit) {
    __shared__ __align__(16) unsigned short ldsA[MTILE * LDS_STRIDE]; // 17,408 B
    __shared__ float shM[8][MTILE];                                   // 2 KB
    __shared__ float shS[8][MTILE];                                   // 2 KB

    const int tid   = threadIdx.x;
    const int lane  = tid & 31;
    const int wave  = tid >> 5;
    const int l15   = lane & 15;
    const int khalf = lane >> 4;             // 0: lanes 0-15, 1: lanes 16-31
    const int vchunk = blockIdx.x;           // 0..124
    const int vbase  = vchunk * VCHUNK;
    const int mbase  = blockIdx.y * MTILE;
    const int cbase  = vbase + wave * 32;    // this wave's 32 private columns
    const int col0   = cbase + l15;
    const int col1   = col0 + 16;

    const unsigned short* wc0 = Wb + (size_t)col0 * D_;
    const unsigned short* wc1 = Wb + (size_t)col1 * D_;

    // A-staging assignment: 4 threads per row, 32 bf16 (64B) each.
    const int srow  = tid >> 2;              // 0..63
    const int spart = tid & 3;               // 0..3
    const unsigned short* sgl = Xb + (size_t)(mbase + srow) * D_ + spart * 32;
    unsigned short* sls = &ldsA[srow * LDS_STRIDE + spart * 32];

    v8f acc[4][2] = {};                      // [M-tile][N-tile] accumulators

    for (int kb0 = 0; kb0 < D_; kb0 += KC) {
        __syncthreads();                     // previous chunk's LDS reads done
        {   // stage A[64][128] bf16 for this K-chunk
            const uint4* g = (const uint4*)(sgl + kb0);
            uint4 q0 = g[0], q1 = g[1], q2 = g[2], q3 = g[3];
            uint4* l = (uint4*)sls;
            l[0] = q0; l[1] = q1; l[2] = q2; l[3] = q3;
        }
        __syncthreads();                     // staged data visible

        for (int kl = 0; kl < KC; kl += 32) {
            // B fragment (32x16 bf16): lane = column, lane-half = contiguous K 0-15/16-31
            const int kB = kb0 + kl + 16 * khalf;
            union { v16bf v; uint4 q[2]; } b0, b1, a[4];
            b0.q[0] = *(const uint4*)(wc0 + kB);
            b0.q[1] = *(const uint4*)(wc0 + kB + 8);
            b1.q[0] = *(const uint4*)(wc1 + kB);
            b1.q[1] = *(const uint4*)(wc1 + kB + 8);
            // A fragment (16x32 bf16): lane-half 0 holds K 0-15, half 1 K 16-31
            const int abase = kl + 16 * khalf;
#pragma unroll
            for (int mi = 0; mi < 4; ++mi) {
                const unsigned short* ap = &ldsA[(16 * mi + l15) * LDS_STRIDE + abase];
                a[mi].q[0] = *(const uint4*)(ap);
                a[mi].q[1] = *(const uint4*)(ap + 8);
            }
#pragma unroll
            for (int mi = 0; mi < 4; ++mi) {
                acc[mi][0] = __builtin_amdgcn_wmma_f32_16x16x32_bf16(false, a[mi].v, false, b0.v, (short)0, acc[mi][0], false, false);
                acc[mi][1] = __builtin_amdgcn_wmma_f32_16x16x32_bf16(false, a[mi].v, false, b1.v, (short)0, acc[mi][1], false, false);
            }
        }
    }

    // Epilogue stage 1: per-wave row (max, sum-exp) over its 32 columns.
    // C/D layout: lane -> column l15; element e -> row (e + 8*khalf) of 16-row tile.
#pragma unroll
    for (int mi = 0; mi < 4; ++mi) {
#pragma unroll
        for (int e = 0; e < 8; ++e) {
            float v0 = acc[mi][0][e], v1 = acc[mi][1][e];
            int rl = 16 * mi + 8 * khalf + e;
            float mx = fmaxf(v0, v1);
#pragma unroll
            for (int off = 1; off < 16; off <<= 1)
                mx = fmaxf(mx, __shfl_xor(mx, off, 32));    // stays within 16-lane half
            float s = __expf(v0 - mx) + __expf(v1 - mx);
#pragma unroll
            for (int off = 1; off < 16; off <<= 1)
                s += __shfl_xor(s, off, 32);
            if (l15 == 0) { shM[wave][rl] = mx; shS[wave][rl] = s; }
            int row = mbase + rl;
            int t = targets[row];
            if (t == col0) tgtLogit[row] = v0;  // unique writer across the whole grid
            if (t == col1) tgtLogit[row] = v1;
        }
    }
    __syncthreads();

    // Epilogue stage 2: merge the 8 wave-partials -> one (max,sum) per (row, vchunk).
    if (tid < MTILE) {
        float M = shM[0][tid], S = shS[0][tid];
#pragma unroll
        for (int w = 1; w < 8; ++w) {
            float mw = shM[w][tid], sw = shS[w][tid];
            float nm = fmaxf(M, mw);
            S = S * __expf(M - nm) + sw * __expf(mw - nm);
            M = nm;
        }
        size_t p = (size_t)(mbase + tid) * gridDim.x + vchunk;
        pMax[p] = M;
        pSum[p] = S;
    }
}

// ---------- kernel 3: merge partials, weighted mean ----------
__global__ __launch_bounds__(1024)
void reduce_kernel(const int* __restrict__ targets,
                   const float* __restrict__ pMax,
                   const float* __restrict__ pSum,
                   const float* __restrict__ tgtLogit,
                   float* __restrict__ out, int nvc) {
    __shared__ float sA[1024];
    __shared__ float sB[1024];
    __shared__ float sCnt[2];
    const int tid = threadIdx.x;

    // per-batch valid counts
    int c0 = 0, c1 = 0;
    for (int s = tid; s < S_; s += 1024) {
        c0 += (targets[s] != IGNORE_INDEX);
        c1 += (targets[S_ + s] != IGNORE_INDEX);
    }
    sA[tid] = (float)c0; sB[tid] = (float)c1;
    __syncthreads();
    for (int st = 512; st > 0; st >>= 1) {
        if (tid < st) { sA[tid] += sA[tid + st]; sB[tid] += sB[tid + st]; }
        __syncthreads();
    }
    if (tid == 0) { sCnt[0] = sA[0]; sCnt[1] = sB[0]; }
    __syncthreads();
    const float wb[2] = { 1.0f / sqrtf(sCnt[0] + 1e-8f),
                          1.0f / sqrtf(sCnt[1] + 1e-8f) };
    __syncthreads();

    float wl = 0.0f, ws = 0.0f;
    for (int r = tid; r < NROWS; r += 1024) {
        int t = targets[r];
        if (t != IGNORE_INDEX) {
            const float* pm = pMax + (size_t)r * nvc;
            const float* ps = pSum + (size_t)r * nvc;
            float M = -__builtin_inff(), S = 0.0f;
            for (int j = 0; j < nvc; ++j) {
                float mj = pm[j], sj = ps[j];
                float nm = fmaxf(M, mj);
                S = S * __expf(M - nm) + sj * __expf(mj - nm);
                M = nm;
            }
            float lse  = M + __logf(S);
            float loss = lse - tgtLogit[r];
            float w = wb[r >> 12];           // r/4096 -> batch index
            wl += loss * w;
            ws += w;
        }
    }
    sA[tid] = wl; sB[tid] = ws;
    __syncthreads();
    for (int st = 512; st > 0; st >>= 1) {
        if (tid < st) { sA[tid] += sA[tid + st]; sB[tid] += sB[tid + st]; }
        __syncthreads();
    }
    if (tid == 0) out[0] = (sB[0] == 0.0f) ? sA[0] : sA[0] / sB[0];
}

extern "C" void kernel_launch(void* const* d_in, const int* in_sizes, int n_in,
                              void* d_out, int out_size, void* d_ws, size_t ws_size,
                              hipStream_t stream) {
    const float* X  = (const float*)d_in[0];   // [B,S,D] f32
    const float* W  = (const float*)d_in[1];   // [V,D]   f32
    const int*   tg = (const int*)d_in[2];     // [B,S]   i32
    float* out = (float*)d_out;

    const long nX = (long)NROWS * D_;          // 8,388,608
    const long nW = (long)V_ * D_;             // 32,768,000

    // workspace layout (256B-aligned blocks): ~86 MiB total
    char* ws = (char*)d_ws;
    size_t off = 0;
    unsigned short* Wb = (unsigned short*)(ws + off); off += ((size_t)nW * 2 + 255) & ~(size_t)255;
    unsigned short* Xb = (unsigned short*)(ws + off); off += ((size_t)nX * 2 + 255) & ~(size_t)255;
    float* pMax = (float*)(ws + off);                 off += ((size_t)NROWS * NVC * 4 + 255) & ~(size_t)255;
    float* pSum = (float*)(ws + off);                 off += ((size_t)NROWS * NVC * 4 + 255) & ~(size_t)255;
    float* tgtL = (float*)(ws + off);                 off += ((size_t)NROWS * 4 + 255) & ~(size_t)255;

    prep_kernel<<<4096, 256, 0, stream>>>(X, W, Xb, Wb, tgtL, nX, nW, NROWS);

    dim3 grid(NVC, NROWS / MTILE);             // 125 x 128 = 16000 blocks
    gemm_partial_kernel<<<grid, 256, 0, stream>>>(Xb, Wb, tg, pMax, pSum, tgtL);

    reduce_kernel<<<1, 1024, 0, stream>>>(tg, pMax, pSum, tgtL, out, NVC);
}
